// LSTMOnlyWithPacking_12309376271042
// MI455X (gfx1250) — matrix operand order repaired and checked by reading, not compiled, not run on hardware
//
#include <hip/hip_runtime.h>

typedef __attribute__((ext_vector_type(16))) _Float16 v16h;
typedef __attribute__((ext_vector_type(8)))  float    v8f;

#define T_SEQ  224
#define HID    128
#define G4     512           // 4*HID gate width
#define NSEQ   1280          // 64*20 sequences
#define MT     16            // sequences per workgroup
#define NINTER 64
#define W1K    (HID * T_SEQ) // 28672

__device__ __forceinline__ float sigf(float v) {
    return 1.0f / (1.0f + __expf(-v));
}
__device__ __forceinline__ float tanhfast(float v) {
    float e = __expf(-2.0f * v);
    return (1.0f - e) / (1.0f + e);
}

// Pre-pass: W1 f32 -> f16 into workspace (3.67 MB, stays L2-resident)
extern "C" __global__ void w1_convert_kernel(const float* __restrict__ W1,
                                             _Float16* __restrict__ W1h, int n) {
    int i = blockIdx.x * 256 + threadIdx.x;
    if (i < n) W1h[i] = (_Float16)W1[i];
}

extern "C" __global__ __launch_bounds__(256, 1)
void lstm_fused_kernel(const float* __restrict__ x,      // [1280][224]
                       const float* __restrict__ Wih,    // [512] (input dim 1)
                       const float* __restrict__ Whh,    // [512][128]
                       const float* __restrict__ bih,
                       const float* __restrict__ bhh,
                       const _Float16* __restrict__ W1h, // [64][28672] f16
                       const float* __restrict__ bias1,  // [64]
                       float* __restrict__ feats)        // [1280][64] out (ws)
{
    __shared__ float    Gbuf[MT * G4];        // 32 KB gate staging
    __shared__ _Float16 hfrag[4 * 32 * 16];   //  4 KB h_t A-fragments
    __shared__ _Float16 hmfrag[4 * 32 * 16];  //  4 KB masked h_t A-fragments
    __shared__ float    xbuf[MT * T_SEQ];     // 14 KB x tile
    __shared__ int      lenb[MT];

    const int tid  = threadIdx.x;
    const int wave = tid >> 5;
    const int lane = tid & 31;
    const int seq0 = blockIdx.x * MT;

    // ---------------- one-time setup ----------------
    for (int i = tid; i < MT * T_SEQ; i += 256) {
        int m = i / T_SEQ, t = i - m * T_SEQ;
        xbuf[i] = x[(size_t)(seq0 + m) * T_SEQ + t];
    }
    for (int i = tid; i < 4 * 32 * 16; i += 256) {
        hfrag[i]  = (_Float16)0.0f;
        hmfrag[i] = (_Float16)0.0f;
    }

    // W_hh^T B-fragments held in REGISTERS for the whole kernel (128 VGPRs).
    // B k order used here (contiguous, same as A): lane 0-15 -> N=lane, K=slot; lane 16-31 -> N=lane-16, K=16+slot.
    // Per lane a fragment is 16 contiguous f32 of one W_hh row -> direct global gather.
    v16h bfr[16];
#pragma unroll
    for (int n = 0; n < 4; ++n) {
#pragma unroll
        for (int kt = 0; kt < 4; ++kt) {
            int N = (wave * 4 + n) * 16 + (lane & 15);
            int K = kt * 32 + ((lane & 16) ? 16 : 0);
            const float* wp = Whh + N * HID + K;
            v16h b;
#pragma unroll
            for (int q = 0; q < 16; ++q) b[q] = (_Float16)wp[q];
            bfr[n * 4 + kt] = b;
        }
    }

    __syncthreads();  // xbuf ready (needed for length scan)

    // ragged lengths: argmax(x==0) semantics; idx 0 (incl. "no zero") -> T, else idx+1
    if (tid < MT) {
        int fz = 0;
        for (int t = 0; t < T_SEQ; ++t) {
            if (xbuf[tid * T_SEQ + t] == 0.0f) { fz = t; break; }
        }
        lenb[tid] = (fz == 0) ? T_SEQ : (fz + 1);
    }
    __syncthreads();

    // fixed thread -> (m, hcol) mapping; cell state + per-item constants in registers
    int em[8], hix[8], lreg[8];
    float creg[8];
    float wi[8], wf[8], wg[8], wo[8];   // W_ih per gate (hoisted)
    float bi[8], bf[8], bg[8], bo[8];   // b_ih + b_hh per gate (hoisted)
#pragma unroll
    for (int e = 0; e < 8; ++e) {
        int idx = e * 256 + tid;
        int m = idx >> 7, hc = idx & 127;
        em[e] = m;
        // A 16x32 fragment layout inverse: (m, k) -> (lane, slot)
        int k  = hc & 31, kt = hc >> 5;
        int ln = m + ((k & 16) ? 16 : 0);
        int s  = k & 15;
        hix[e] = (kt * 32 + ln) * 16 + s;
        creg[e] = 0.0f;
        wi[e] = Wih[hc];        bi[e] = bih[hc]        + bhh[hc];
        wf[e] = Wih[128 + hc];  bf[e] = bih[128 + hc]  + bhh[128 + hc];
        wg[e] = Wih[256 + hc];  bg[e] = bih[256 + hc]  + bhh[256 + hc];
        wo[e] = Wih[384 + hc];  bo[e] = bih[384 + hc]  + bhh[384 + hc];
        lreg[e] = lenb[m];
    }

    // fused-W1 GEMM assignment: wave -> (N-tile of 4, K-half of 2)
    const int fnt = wave & 3;
    const int fkh = wave >> 2;
    const _Float16* w1base = W1h + (size_t)(fnt * 16 + (lane & 15)) * W1K
                                 + (fkh * 2) * 32 + ((lane & 16) ? 16 : 0);
    v8f facc = {0.f, 0.f, 0.f, 0.f, 0.f, 0.f, 0.f, 0.f};
    const v8f vzero = {0.f, 0.f, 0.f, 0.f, 0.f, 0.f, 0.f, 0.f};

    // ---------------- time loop ----------------
    for (int t = 0; t < T_SEQ; ++t) {
        // G = h_{t-1} @ W_hh^T : load 4 A-fragments, then 16 register-fed WMMAs
        v16h a[4];
#pragma unroll
        for (int kt = 0; kt < 4; ++kt)
            a[kt] = *(const v16h*)(hfrag + (kt * 32 + lane) * 16);

        v8f acc[4];
#pragma unroll
        for (int n = 0; n < 4; ++n) acc[n] = vzero;
#pragma unroll
        for (int kt = 0; kt < 4; ++kt) {
#pragma unroll
            for (int n = 0; n < 4; ++n) {
                acc[n] = __builtin_amdgcn_wmma_f32_16x16x32_f16(
                    false, a[kt], false, bfr[n * 4 + kt], (short)0, acc[n], false, false);
            }
        }
        // stage gate tiles to LDS (D layout: lane<16 -> M=j, lane>=16 -> M=j+8)
#pragma unroll
        for (int n = 0; n < 4; ++n) {
            int col = (wave * 4 + n) * 16 + (lane & 15);
            int rb  = (lane & 16) ? 8 : 0;
            const float* ap = (const float*)&acc[n];
#pragma unroll
            for (int j = 0; j < 8; ++j) Gbuf[(rb + j) * G4 + col] = ap[j];
        }
        if (t + 1 < T_SEQ)
            __builtin_prefetch(w1base + (size_t)(t + 1) * HID, 0, 3);
        __syncthreads();

        // elementwise LSTM cell (gx = x*W_ih + b folded in; rank-1 input)
#pragma unroll
        for (int e = 0; e < 8; ++e) {
            int m = em[e];
            int hc = ((e * 256 + tid) & 127);
            float xv = xbuf[m * T_SEQ + t];
            const float* gr = Gbuf + m * G4;
            float gi = gr[hc]       + xv * wi[e] + bi[e];
            float gf = gr[128 + hc] + xv * wf[e] + bf[e];
            float gg = gr[256 + hc] + xv * wg[e] + bg[e];
            float go = gr[384 + hc] + xv * wo[e] + bo[e];
            float c  = sigf(gf) * creg[e] + sigf(gi) * tanhfast(gg);
            float h  = sigf(go) * tanhfast(c);
            creg[e]  = c;
            _Float16 hh = (_Float16)h;
            hfrag[hix[e]]  = hh;                                // feeds recurrence
            hmfrag[hix[e]] = (t < lreg[e]) ? hh : (_Float16)0.0f; // feeds W1 GEMM
        }
        __syncthreads();

        // fused feats GEMM: facc += h_t(masked) @ W1[:, t*128:(t+1)*128]^T (this wave's K-half)
        {
            const _Float16* wp = w1base + (size_t)t * HID;
            v16h a0 = *(const v16h*)(hmfrag + ((fkh * 2 + 0) * 32 + lane) * 16);
            v16h b0 = *(const v16h*)wp;
            facc = __builtin_amdgcn_wmma_f32_16x16x32_f16(
                false, a0, false, b0, (short)0, facc, false, false);
            v16h a1 = *(const v16h*)(hmfrag + ((fkh * 2 + 1) * 32 + lane) * 16);
            v16h b1 = *(const v16h*)(wp + 32);
            facc = __builtin_amdgcn_wmma_f32_16x16x32_f16(
                false, a1, false, b1, (short)0, facc, false, false);
        }
    }

    // ---------------- combine feats K-halves, add b1, write out ----------------
    float* pbuf = Gbuf; // reuse staging buffer: [2 khalf][16 m][64 n]
    {
        int col = fnt * 16 + (lane & 15);
        int rb  = (lane & 16) ? 8 : 0;
        const float* fp = (const float*)&facc;
#pragma unroll
        for (int j = 0; j < 8; ++j)
            pbuf[fkh * (MT * NINTER) + (rb + j) * NINTER + col] = fp[j];
    }
    __syncthreads();
    for (int idx = tid; idx < MT * NINTER; idx += 256) {
        float f = pbuf[idx] + pbuf[MT * NINTER + idx] + bias1[idx & 63];
        feats[(size_t)(seq0 + (idx >> 6)) * NINTER + (idx & 63)] = f;
    }
}

// head: out[b][j] = b2[j] + sum_k feats_flat[b][k] * W2[j][k]   (k = s*64+i matches layout)
extern "C" __global__ __launch_bounds__(256)
void head_kernel(const float* __restrict__ feats, const float* __restrict__ W2,
                 const float* __restrict__ b2, float* __restrict__ out) {
    __shared__ float red0[256], red1[256];
    int b = blockIdx.x, tid = threadIdx.x;
    float a0 = 0.f, a1 = 0.f;
    for (int k = tid; k < 1280; k += 256) {
        float fv = feats[b * 1280 + k];
        a0 += fv * W2[k];
        a1 += fv * W2[1280 + k];
    }
    red0[tid] = a0; red1[tid] = a1;
    __syncthreads();
    for (int s = 128; s > 0; s >>= 1) {
        if (tid < s) { red0[tid] += red0[tid + s]; red1[tid] += red1[tid + s]; }
        __syncthreads();
    }
    if (tid == 0) {
        out[b * 2 + 0] = red0[0] + b2[0];
        out[b * 2 + 1] = red1[0] + b2[1];
    }
}

extern "C" void kernel_launch(void* const* d_in, const int* in_sizes, int n_in,
                              void* d_out, int out_size, void* d_ws, size_t ws_size,
                              hipStream_t stream) {
    (void)in_sizes; (void)n_in; (void)out_size; (void)ws_size;
    const float* x    = (const float*)d_in[0];
    // d_in[1] metadata: unused by the reference
    const float* Wih  = (const float*)d_in[2];
    const float* Whh  = (const float*)d_in[3];
    const float* bih  = (const float*)d_in[4];
    const float* bhh  = (const float*)d_in[5];
    const float* W1   = (const float*)d_in[6];
    const float* b1   = (const float*)d_in[7];
    const float* W2   = (const float*)d_in[8];
    const float* b2   = (const float*)d_in[9];

    _Float16* W1h   = (_Float16*)d_ws;                                   // 3,670,016 B
    float*    feats = (float*)((char*)d_ws + (size_t)NINTER * W1K * 2);  //   327,680 B

    int n1 = NINTER * W1K;
    w1_convert_kernel<<<(n1 + 255) / 256, 256, 0, stream>>>(W1, W1h, n1);

    lstm_fused_kernel<<<NSEQ / MT, 256, 0, stream>>>(
        x, Wih, Whh, bih, bhh, W1h, b1, feats);

    head_kernel<<<64, 256, 0, stream>>>(feats, W2, b2, (float*)d_out);
}
